// MLA_40561671143651
// MI455X (gfx1250) — matrix operand order repaired and checked by reading, not compiled, not run on hardware
//
#include <hip/hip_runtime.h>
#include <cstdint>

#define S_LEN 2048
#define HIDD  2048
#define NH    16
#define DNN   128
#define DRR   64
#define DVV   128
#define QLAT  1536
#define KVLAT 512
#define ACOLS (QLAT + KVLAT + DRR)   // 2112
#define QD    (DNN + DRR)            // 192
#define KVD   (DNN + DVV)            // 256

typedef __bf16 bf16;
typedef _Float16 f16;
typedef __attribute__((ext_vector_type(16))) __bf16  bf16x16;
typedef __attribute__((ext_vector_type(8)))  __bf16  bf16x8;
typedef __attribute__((ext_vector_type(16))) _Float16 f16x16;
typedef __attribute__((ext_vector_type(8)))  _Float16 f16x8;
typedef __attribute__((ext_vector_type(8)))  float    f32x8;

static __device__ __forceinline__ bf16 f2bf(float f) {
  union { float f; uint32_t u; } v; v.f = f;
  uint32_t r = v.u + 0x7FFFu + ((v.u >> 16) & 1u);   // round-to-nearest-even
  uint16_t h = (uint16_t)(r >> 16);
  return __builtin_bit_cast(bf16, h);
}

static __device__ __forceinline__ f32x8 wmma_bf16(bf16x16 a, bf16x16 b, f32x8 c) {
  return __builtin_amdgcn_wmma_f32_16x16x32_bf16(false, a, false, b, (short)0, c,
                                                 false, false);
}
static __device__ __forceinline__ f32x8 wmma_f16(f16x16 a, f16x16 b, f32x8 c) {
  return __builtin_amdgcn_wmma_f32_16x16x32_f16(false, a, false, b, (short)0, c,
                                                false, false);
}

// A-matrix per-lane load (16-bit, 16x32 tile, row-major source, K-contiguous):
// lanes 0-15: elems 0..15 = K+0..15
// lanes16-31: elems 0..15 = K+16..31
static __device__ __forceinline__ bf16x16 load_a16(const bf16* p, int hi) {
  const bf16x8* q = (const bf16x8*)(p + (hi << 4));
  bf16x8 lo = q[0];
  bf16x8 hv = q[1];   // +8 halves
  return __builtin_shufflevector(lo, hv, 0,1,2,3,4,5,6,7,8,9,10,11,12,13,14,15);
}
static __device__ __forceinline__ f16x16 load_a16h(const f16* p, int hi) {
  const f16x8* q = (const f16x8*)(p + (hi << 4));
  f16x8 lo = q[0];
  f16x8 hv = q[1];
  return __builtin_shufflevector(lo, hv, 0,1,2,3,4,5,6,7,8,9,10,11,12,13,14,15);
}

// ---------- conversion kernels ----------
__global__ void cvt_f32_bf16(const float* __restrict__ in, bf16* __restrict__ out,
                             int n) {
  for (int i = blockIdx.x * blockDim.x + threadIdx.x; i < n;
       i += gridDim.x * blockDim.x)
    out[i] = f2bf(in[i]);
}

// in: [R][C] row-major f32  ->  out: [C][R] bf16 (transposed)
__global__ void tcvt_f32_bf16(const float* __restrict__ in, bf16* __restrict__ out,
                              int R, int C) {
  int total = R * C;
  for (int i = blockIdx.x * blockDim.x + threadIdx.x; i < total;
       i += gridDim.x * blockDim.x) {
    int c = i / R, r = i - c * R;
    out[i] = f2bf(in[(size_t)r * C + c]);
  }
}

// ---------- GEMM: C[M][N] f32 = A[M][K] bf16 @ Bt[N][K] bf16 ----------
// grid = (M/64, N/64), block = 32 : one wave computes a 64x64 tile
// (4x4 WMMA accumulators, ~32 flops/byte from L2)
__global__ __launch_bounds__(32) void gemm_bf16(
    float* __restrict__ C, const bf16* __restrict__ A, const bf16* __restrict__ Bt,
    int M, int N, int K) {
  const int lane = threadIdx.x & 31;
  const int lm = lane & 15;
  const int hi = lane >> 4;
  const int m0 = blockIdx.x << 6;
  const int n0 = blockIdx.y << 6;
  const bf16* arow = A + (size_t)(m0 + lm) * K;
  const bf16* brow = Bt + (size_t)(n0 + lm) * K + (hi << 4);
  const size_t step16 = (size_t)K << 4;   // 16 rows
  f32x8 acc[4][4] = {};
  for (int k0 = 0; k0 < K; k0 += 32) {
    if (k0 + 256 < K) {
#pragma unroll
      for (int i = 0; i < 4; ++i) {
        // locality=3 -> WGP-scope prefetch (fills all cache levels; consumed by
        // this same wave ~8 K-steps later)
        __builtin_prefetch(arow + i * step16 + k0 + 256, 0, 3);
        __builtin_prefetch(brow + i * step16 + k0 + 256, 0, 3);
      }
    }
    bf16x16 a[4];
#pragma unroll
    for (int i = 0; i < 4; ++i) a[i] = load_a16(arow + i * step16 + k0, hi);
#pragma unroll
    for (int j = 0; j < 4; ++j) {
      bf16x16 b = *(const bf16x16*)(brow + j * step16 + k0);
#pragma unroll
      for (int i = 0; i < 4; ++i) acc[i][j] = wmma_bf16(a[i], b, acc[i][j]);
    }
  }
#pragma unroll
  for (int i = 0; i < 4; ++i) {
    float* crow = C + (size_t)(m0 + i * 16 + hi * 8) * N + n0 + lm;
#pragma unroll
    for (int r = 0; r < 8; ++r)
#pragma unroll
      for (int j = 0; j < 4; ++j)
        crow[(size_t)r * N + j * 16] = acc[i][j][r];
  }
}

// ---------- rmsnorm(q,kv) + k_pe RoPE broadcast into k_full ----------
// grid = S, block = 256
__global__ __launch_bounds__(256) void rmsnorm_rope(
    const float* __restrict__ a, const int* __restrict__ pos,
    const float* __restrict__ gq, const float* __restrict__ gkv,
    bf16* __restrict__ qlat, bf16* __restrict__ ckv, bf16* __restrict__ kfull) {
  const int s = blockIdx.x, tid = threadIdx.x;
  const float* row = a + (size_t)s * ACOLS;
  __shared__ float red[256];
  __shared__ float pe[DRR];

  // q-latent rmsnorm
  float ss = 0.f;
  for (int j = tid; j < QLAT; j += 256) { float v = row[j]; ss += v * v; }
  red[tid] = ss; __syncthreads();
  for (int off = 128; off > 0; off >>= 1) {
    if (tid < off) red[tid] += red[tid + off];
    __syncthreads();
  }
  float rq = rsqrtf(red[0] / (float)QLAT + 1e-6f);
  __syncthreads();
  for (int j = tid; j < QLAT; j += 256)
    qlat[(size_t)s * QLAT + j] = f2bf(row[j] * rq * gq[j]);

  // kv-latent rmsnorm
  float s2 = 0.f;
  for (int j = tid; j < KVLAT; j += 256) { float v = row[QLAT + j]; s2 += v * v; }
  red[tid] = s2; __syncthreads();
  for (int off = 128; off > 0; off >>= 1) {
    if (tid < off) red[tid] += red[tid + off];
    __syncthreads();
  }
  float rkv = rsqrtf(red[0] / (float)KVLAT + 1e-6f);
  __syncthreads();
  for (int j = tid; j < KVLAT; j += 256)
    ckv[(size_t)s * KVLAT + j] = f2bf(row[QLAT + j] * rkv * gkv[j]);

  // k_pe RoPE (shared across heads) -> broadcast into k_full[h][s][128..191]
  if (tid < 32) {
    float p = (float)pos[s];
    float inv = powf(10000.f, -(float)(2 * tid) / 64.f);
    float ang = p * inv;
    float c = cosf(ang), sn = sinf(ang);
    float x1 = row[QLAT + KVLAT + tid];
    float x2 = row[QLAT + KVLAT + 32 + tid];
    pe[tid]      = x1 * c - x2 * sn;
    pe[tid + 32] = x2 * c + x1 * sn;
  }
  __syncthreads();
  for (int idx = tid; idx < NH * DRR; idx += 256) {
    int h = idx >> 6, d = idx & 63;
    kfull[((size_t)h * S_LEN + s) * QD + DNN + d] = f2bf(pe[d]);
  }
}

// ---------- q post-process: rope q_pe, pack q_full[h][s][192] bf16 ----------
__global__ __launch_bounds__(256) void q_post(
    const float* __restrict__ q, const int* __restrict__ pos,
    bf16* __restrict__ qfull) {
  const int s = blockIdx.x, tid = threadIdx.x;
  __shared__ float cs[32], sn[32];
  if (tid < 32) {
    float p = (float)pos[s];
    float inv = powf(10000.f, -(float)(2 * tid) / 64.f);
    cs[tid] = cosf(p * inv);
    sn[tid] = sinf(p * inv);
  }
  __syncthreads();
  const float* row = q + (size_t)s * (NH * QD);
  for (int idx = tid; idx < NH * QD; idx += 256) {
    int h = idx / QD, d = idx - h * QD;
    float val;
    if (d < DNN) {
      val = row[h * QD + d];
    } else {
      int j = d - DNN;
      if (j < 32) {
        float x1 = row[h * QD + DNN + j], x2 = row[h * QD + DNN + 32 + j];
        val = x1 * cs[j] - x2 * sn[j];
      } else {
        j -= 32;
        float x1 = row[h * QD + DNN + j], x2 = row[h * QD + DNN + 32 + j];
        val = x2 * cs[j] + x1 * sn[j];
      }
    }
    qfull[((size_t)h * S_LEN + s) * QD + d] = f2bf(val);
  }
}

// ---------- kv post-process: k_nope -> k_full (bf16), v -> vt[h][128][S] f16 ----
__global__ __launch_bounds__(256) void kv_post(
    const float* __restrict__ kv, bf16* __restrict__ kfull, f16* __restrict__ vt) {
  const int s = blockIdx.x, tid = threadIdx.x;
  const float* row = kv + (size_t)s * (NH * KVD);
  for (int idx = tid; idx < NH * KVD; idx += 256) {
    int h = idx >> 8, d = idx & 255;
    float v = row[idx];
    if (d < DNN)
      kfull[((size_t)h * S_LEN + s) * QD + d] = f2bf(v);
    else
      vt[((size_t)h * DVV + (d - DNN)) * S_LEN + s] = (f16)v;
  }
}

// ---------- fused attention: per (head, 16-query tile) workgroup ----------
// grid = (S/16, H), block = 128 (4 waves)
// dyn LDS: f32 scores [16][S] + f16 probs [16][S] + reduction scratch (~193 KB)
__global__ __launch_bounds__(128) void mla_attn(
    const bf16* __restrict__ qfull, const bf16* __restrict__ kfull,
    const f16* __restrict__ vt, bf16* __restrict__ attn) {
  extern __shared__ char smem_raw[];
  float* sc     = (float*)smem_raw;                       // [16][S_LEN] raw scores
  f16*   pf     = (f16*)(smem_raw + 16 * S_LEN * 4);      // [16][S_LEN] exp probs
  float* red    = (float*)(smem_raw + 16 * S_LEN * 6);    // [16][8]
  float* rowmax = red + 128;                              // [16]
  float* rowinv = rowmax + 16;                            // [16]

  const int h = blockIdx.y;
  const int qbase = blockIdx.x << 4;
  const int tid = threadIdx.x;
  const int wave = tid >> 5, lane = tid & 31;
  const int lm = lane & 15, hi = lane >> 4;
  const float scale = 0.07216878364870323f;   // 1/sqrt(192)

  // Q tile (16 x 192) in registers, 6 K-chunks of 32
  const bf16* qrow = qfull + ((size_t)h * S_LEN + qbase + lm) * QD;
  bf16x16 aq[6];
#pragma unroll
  for (int kb = 0; kb < 6; ++kb) aq[kb] = load_a16(qrow + kb * 32, hi);

  // scores = (q_nope|q_pe) . (k_nope|k_pe)^T, scaled + causal-masked
  for (int tt = wave; tt < S_LEN / 16; tt += 4) {
    const bf16* krow = kfull + ((size_t)h * S_LEN + tt * 16 + lm) * QD + (hi << 4);
    f32x8 c = {};
#pragma unroll
    for (int kb = 0; kb < 6; ++kb) {
      bf16x16 b = *(const bf16x16*)(krow + kb * 32);
      c = wmma_bf16(aq[kb], b, c);
    }
    const int tg = (tt << 4) + lm;
#pragma unroll
    for (int r = 0; r < 8; ++r) {
      int M = r + hi * 8;
      float v = c[r] * scale;
      if (tg > qbase + M) v = -1e30f;
      sc[M * S_LEN + tg] = v;
    }
  }
  __syncthreads();

  // exact full-row softmax: f32 max/sum, store unnormalized exp as f16
  {
    const int row = tid >> 3, sub = tid & 7;
    const int t0 = sub * (S_LEN / 8);
    float mx = -1e30f;
    for (int t = 0; t < S_LEN / 8; ++t) mx = fmaxf(mx, sc[row * S_LEN + t0 + t]);
    red[row * 8 + sub] = mx;
    __syncthreads();
    if (sub == 0) {
      float m = red[row * 8];
      for (int i = 1; i < 8; ++i) m = fmaxf(m, red[row * 8 + i]);
      rowmax[row] = m;
    }
    __syncthreads();
    float m = rowmax[row];
    float sum = 0.f;
    for (int t = 0; t < S_LEN / 8; ++t) {
      float e = __expf(sc[row * S_LEN + t0 + t] - m);
      pf[row * S_LEN + t0 + t] = (f16)e;
      sum += e;
    }
    red[row * 8 + sub] = sum;
    __syncthreads();
    if (sub == 0) {
      float s2 = 0.f;
      for (int i = 0; i < 8; ++i) s2 += red[row * 8 + i];
      rowinv[row] = 1.f / s2;
    }
  }
  __syncthreads();

  // out = P . V (f16 WMMA, K = S_LEN); each wave owns two 16-wide d-tiles
#pragma unroll
  for (int dti = 0; dti < 2; ++dti) {
    const int d0 = (wave * 2 + dti) << 4;
    const f16* vrow = vt + ((size_t)h * DVV + d0 + lm) * S_LEN + (hi << 4);
    f32x8 c = {};
    for (int t0 = 0; t0 < S_LEN; t0 += 32) {
      f16x16 a = load_a16h(pf + lm * S_LEN + t0, hi);   // 2x ds_load_b128
      f16x16 b = *(const f16x16*)(vrow + t0);
      c = wmma_f16(a, b, c);
    }
#pragma unroll
    for (int r = 0; r < 8; ++r) {
      int M = r + hi * 8;
      attn[(size_t)(qbase + M) * (NH * DVV) + h * DVV + d0 + lm] =
          f2bf(c[r] * rowinv[M]);
    }
  }
}

// ------------------------------------------------------------------
extern "C" void kernel_launch(void* const* d_in, const int* in_sizes, int n_in,
                              void* d_out, int out_size, void* d_ws, size_t ws_size,
                              hipStream_t stream) {
  const float* hs   = (const float*)d_in[0];
  const int*   pos  = (const int*)d_in[1];
  const float* wA   = (const float*)d_in[2];
  const float* gq   = (const float*)d_in[3];
  const float* gkv  = (const float*)d_in[4];
  const float* wQb  = (const float*)d_in[5];
  const float* wKVb = (const float*)d_in[6];
  const float* wO   = (const float*)d_in[7];
  float* out = (float*)d_out;
  (void)in_sizes; (void)n_in; (void)out_size; (void)ws_size;

  char* p = (char*)d_ws;
  auto take = [&](size_t bytes) {
    char* r = p; p += (bytes + 255) & ~(size_t)255; return r;
  };
  bf16*  hs_b   = (bf16*) take((size_t)S_LEN * HIDD * 2);
  bf16*  wA_t   = (bf16*) take((size_t)ACOLS * HIDD * 2);
  bf16*  wQb_t  = (bf16*) take((size_t)(NH * QD) * QLAT * 2);
  bf16*  wKVb_t = (bf16*) take((size_t)(NH * KVD) * KVLAT * 2);
  bf16*  wO_t   = (bf16*) take((size_t)HIDD * (NH * DVV) * 2);
  float* a_f    = (float*)take((size_t)S_LEN * ACOLS * 4);
  bf16*  qlat_b = (bf16*) take((size_t)S_LEN * QLAT * 2);
  bf16*  ckv_b  = (bf16*) take((size_t)S_LEN * KVLAT * 2);
  float* q_f    = (float*)take((size_t)S_LEN * NH * QD * 4);
  float* kv_f   = (float*)take((size_t)S_LEN * NH * KVD * 4);
  bf16*  q_full = (bf16*) take((size_t)NH * S_LEN * QD * 2);
  bf16*  k_full = (bf16*) take((size_t)NH * S_LEN * QD * 2);
  f16*   vt     = (f16*)  take((size_t)NH * DVV * S_LEN * 2);
  bf16*  attn_b = (bf16*) take((size_t)S_LEN * NH * DVV * 2);

  // 1) precision conversion (weights transposed -> K-contiguous B operands)
  cvt_f32_bf16 <<<8192, 256, 0, stream>>>(hs, hs_b, S_LEN * HIDD);
  tcvt_f32_bf16<<<8192, 256, 0, stream>>>(wA,   wA_t,   HIDD,  ACOLS);
  tcvt_f32_bf16<<<8192, 256, 0, stream>>>(wQb,  wQb_t,  QLAT,  NH * QD);
  tcvt_f32_bf16<<<8192, 256, 0, stream>>>(wKVb, wKVb_t, KVLAT, NH * KVD);
  tcvt_f32_bf16<<<8192, 256, 0, stream>>>(wO,   wO_t,   NH * DVV, HIDD);

  // 2) a = hidden @ w_fused_a
  gemm_bf16<<<dim3(S_LEN / 64, ACOLS / 64), 32, 0, stream>>>(
      a_f, hs_b, wA_t, S_LEN, ACOLS, HIDD);

  // 3) rmsnorm + k_pe rope
  rmsnorm_rope<<<S_LEN, 256, 0, stream>>>(a_f, pos, gq, gkv, qlat_b, ckv_b, k_full);

  // 4) q = q_lat @ w_q_b ; kv = ckv @ w_kv_b
  gemm_bf16<<<dim3(S_LEN / 64, (NH * QD) / 64), 32, 0, stream>>>(
      q_f, qlat_b, wQb_t, S_LEN, NH * QD, QLAT);
  gemm_bf16<<<dim3(S_LEN / 64, (NH * KVD) / 64), 32, 0, stream>>>(
      kv_f, ckv_b, wKVb_t, S_LEN, NH * KVD, KVLAT);

  // 5) pack q_full / k_full / vt
  q_post <<<S_LEN, 256, 0, stream>>>(q_f, pos, q_full);
  kv_post<<<S_LEN, 256, 0, stream>>>(kv_f, k_full, vt);

  // 6) fused attention (193 KB dynamic LDS per workgroup; 320 KB/WGP available)
  const int smem_bytes = 16 * S_LEN * 6 + (128 + 16 + 16) * (int)sizeof(float);
  hipFuncSetAttribute((const void*)mla_attn,
                      hipFuncAttributeMaxDynamicSharedMemorySize, smem_bytes);
  mla_attn<<<dim3(S_LEN / 16, NH), 128, smem_bytes, stream>>>(
      q_full, k_full, vt, attn_b);

  // 7) out = attn @ w_o
  gemm_bf16<<<dim3(S_LEN / 64, HIDD / 64), 32, 0, stream>>>(
      out, attn_b, wO_t, S_LEN, HIDD, NH * DVV);
}
